// SpikeFP16Adder_43860206027344
// MI455X (gfx1250) — matrix operand order repaired and checked by reading, not compiled, not run on hardware
//
#include <hip/hip_runtime.h>

// CDNA5 / gfx1250, wave32.
typedef _Float16 v16h __attribute__((ext_vector_type(16)));
typedef float    v8f  __attribute__((ext_vector_type(8)));
typedef float    f32x4 __attribute__((ext_vector_type(4)));

// Convert 16 spike floats (exact 0.0/1.0) to the WMMA B-operand f16 vector.
// B operand (32x16, f16), contiguous k order used here (same as A): lane n<16 -> column n, rows K=0..15 (pair 2v,2v+1 in VGPR v);
// lane n>=16 -> column n-16, rows K=16..31. Either way: just the 16 bits in order.
__device__ __forceinline__ v16h pack_bits_f16(f32x4 q0, f32x4 q1, f32x4 q2, f32x4 q3) {
    v16h r;
    r[0]  = (_Float16)q0.x; r[1]  = (_Float16)q0.y; r[2]  = (_Float16)q0.z; r[3]  = (_Float16)q0.w;
    r[4]  = (_Float16)q1.x; r[5]  = (_Float16)q1.y; r[6]  = (_Float16)q1.z; r[7]  = (_Float16)q1.w;
    r[8]  = (_Float16)q2.x; r[9]  = (_Float16)q2.y; r[10] = (_Float16)q2.z; r[11] = (_Float16)q2.w;
    r[12] = (_Float16)q3.x; r[13] = (_Float16)q3.y; r[14] = (_Float16)q3.z; r[15] = (_Float16)q3.w;
    return r;
}

__global__ __launch_bounds__(256) void spike_fp16_add_kernel(
    const float* __restrict__ A, const float* __restrict__ B,
    float* __restrict__ out, int n)
{
    const int  lane    = threadIdx.x & 31;
    const bool hiHalf  = lane >= 16;
    const long long idx      = (long long)blockIdx.x * blockDim.x + threadIdx.x; // element index
    const long long waveBase = idx - lane;

    // -------- Constant A-operand (weight) matrices, built once per thread. --------
    // A-matrix 16x32 contiguous k order used here (matches B): element e (pair e/2 in VGPR e/2) maps to
//   K = e + (hiHalf ? 16 : 0)
    // Every row M carries the same weight vector w[k] = 2^(15-k) (MSB-first), so every
    // D column equals the packed uint16 of that column's bits.
    // wlo: weights at K=0..15 (consumes B rows from lanes 0-15); whi: weights at K=16..31.
    v16h wlo, whi;
    #pragma unroll
    for (int e = 0; e < 16; ++e) {
        const int K = e + (hiHalf ? 16 : 0);
        const _Float16 w = (_Float16)(float)(1u << (15 - (K & 15))); // exact power of two in f16
        wlo[e] = (K < 16) ? w : (_Float16)0.0f;
        whi[e] = (K < 16) ? (_Float16)0.0f : w;
    }

    if (waveBase + 32 <= (long long)n) {
        // ---------------- Full wave: WMMA pack path (EXEC all ones). ----------------
        const f32x4* pa = (const f32x4*)(A + (size_t)idx * 16);
        const f32x4* pb = (const f32x4*)(B + (size_t)idx * 16);
        // Streamed once, >L2-sized working set: non-temporal loads.
        f32x4 a0 = __builtin_nontemporal_load(pa + 0);
        f32x4 a1 = __builtin_nontemporal_load(pa + 1);
        f32x4 a2 = __builtin_nontemporal_load(pa + 2);
        f32x4 a3 = __builtin_nontemporal_load(pa + 3);
        f32x4 b0 = __builtin_nontemporal_load(pb + 0);
        f32x4 b1 = __builtin_nontemporal_load(pb + 1);
        f32x4 b2 = __builtin_nontemporal_load(pb + 2);
        f32x4 b3 = __builtin_nontemporal_load(pb + 3);

        const v16h ba = pack_bits_f16(a0, a1, a2, a3);
        const v16h bb = pack_bits_f16(b0, b1, b2, b3);

        const v8f c = {};
        // Pack 32 operand-A numbers: lanes 0-15 read via wlo, lanes 16-31 via whi.
        v8f dAlo = __builtin_amdgcn_wmma_f32_16x16x32_f16(false, wlo, false, ba, (short)0, c, false, false);
        v8f dAhi = __builtin_amdgcn_wmma_f32_16x16x32_f16(false, whi, false, ba, (short)0, c, false, false);
        v8f dBlo = __builtin_amdgcn_wmma_f32_16x16x32_f16(false, wlo, false, bb, (short)0, c, false, false);
        v8f dBhi = __builtin_amdgcn_wmma_f32_16x16x32_f16(false, whi, false, bb, (short)0, c, false, false);

        // D rows are identical; VGPR0 of lane L holds lane L's own packed value (exact integer).
        const unsigned ua = (unsigned)(hiHalf ? dAhi[0] : dAlo[0]);
        const unsigned ub = (unsigned)(hiHalf ? dBhi[0] : dBlo[0]);

        // IEEE fp16 add, RNE, denorms preserved (hardware default modes).
        const _Float16 fa = __builtin_bit_cast(_Float16, (unsigned short)ua);
        const _Float16 fb = __builtin_bit_cast(_Float16, (unsigned short)ub);
        const _Float16 fs = fa + fb;
        const unsigned  us = (unsigned)__builtin_bit_cast(unsigned short, fs);

        // Unpack MSB-first to 16 spike floats; non-temporal b128 stores.
        f32x4* po = (f32x4*)(out + (size_t)idx * 16);
        #pragma unroll
        for (int j = 0; j < 4; ++j) {
            f32x4 o;
            o.x = (float)((us >> (15 - (4 * j + 0))) & 1u);
            o.y = (float)((us >> (15 - (4 * j + 1))) & 1u);
            o.z = (float)((us >> (15 - (4 * j + 2))) & 1u);
            o.w = (float)((us >> (15 - (4 * j + 3))) & 1u);
            __builtin_nontemporal_store(o, po + j);
        }
    } else if (idx < (long long)n) {
        // ---------------- Tail: scalar FMA-pack path (no WMMA, EXEC may be partial). ----------------
        const float* pa = A + (size_t)idx * 16;
        const float* pb = B + (size_t)idx * 16;
        float accA = 0.0f, accB = 0.0f;
        #pragma unroll
        for (int i = 0; i < 16; ++i) {
            const float w = (float)(1u << (15 - i));
            accA = fmaf(pa[i], w, accA);   // exact: sums of distinct powers of two < 2^24
            accB = fmaf(pb[i], w, accB);
        }
        const _Float16 fa = __builtin_bit_cast(_Float16, (unsigned short)(unsigned)accA);
        const _Float16 fb = __builtin_bit_cast(_Float16, (unsigned short)(unsigned)accB);
        const _Float16 fs = fa + fb;
        const unsigned us = (unsigned)__builtin_bit_cast(unsigned short, fs);
        float* po = out + (size_t)idx * 16;
        #pragma unroll
        for (int i = 0; i < 16; ++i)
            po[i] = (float)((us >> (15 - i)) & 1u);
    }
}

extern "C" void kernel_launch(void* const* d_in, const int* in_sizes, int n_in,
                              void* d_out, int out_size, void* d_ws, size_t ws_size,
                              hipStream_t stream) {
    (void)n_in; (void)d_ws; (void)ws_size; (void)out_size;
    const float* A = (const float*)d_in[0];
    const float* B = (const float*)d_in[1];
    float* out = (float*)d_out;
    const int n = in_sizes[0] / 16;              // number of fp16 elements (2,097,152)
    const int threads = 256;                     // 8 wave32 waves per block
    const int blocks  = (n + threads - 1) / threads;
    hipLaunchKernelGGL(spike_fp16_add_kernel, dim3(blocks), dim3(threads), 0, stream,
                       A, B, out, n);
}
